// HeteroGAT_90778428768393
// MI455X (gfx1250) — matrix operand order repaired and checked by reading, not compiled, not run on hardware
//
#include <hip/hip_runtime.h>
#include <hip/hip_bf16.h>

typedef _Float16 v16h __attribute__((ext_vector_type(16)));
typedef _Float16 v8h  __attribute__((ext_vector_type(8)));
typedef float    v8f  __attribute__((ext_vector_type(8)));

constexpr int GH  = 4;    // heads
constexpr int GC  = 128;  // channels per head
constexpr int GHC = GH * GC; // 512
constexpr float NEG_SLOPE = 0.2f;

// ---------------------------------------------------------------------------
// Utility device functions
// ---------------------------------------------------------------------------
__device__ inline float waveReduceSum(float v) {
#pragma unroll
    for (int off = 16; off > 0; off >>= 1)
        v += __shfl_xor(v, off, 32);
    return v;
}

__device__ inline void atomicMaxF(float* addr, float val) {
    int* ia = (int*)addr;
    int cur = __float_as_int(*addr);
    while (__int_as_float(cur) < val) {
        int prev = atomicCAS(ia, cur, __float_as_int(val));
        if (prev == cur) break;
        cur = prev;
    }
}

// ---------------------------------------------------------------------------
// fp32 -> fp16 pack (float4 per thread for bandwidth)
// ---------------------------------------------------------------------------
__global__ void f32_to_f16(const float* __restrict__ in, _Float16* __restrict__ out, long n4) {
    const long i = (long)blockIdx.x * blockDim.x + threadIdx.x;
    if (i >= n4) return;
    const float4 v = ((const float4*)in)[i];
    v8h* o = (v8h*)out;  // write 4 halves via vector of 8? use scalar quad instead
    _Float16* p = out + i * 4;
    p[0] = (_Float16)v.x; p[1] = (_Float16)v.y;
    p[2] = (_Float16)v.z; p[3] = (_Float16)v.w;
    (void)o;
}

// ---------------------------------------------------------------------------
// WMMA GEMM:  C[M,N] = A[M,K] @ W[N,K]^T + bias[N]   (torch Linear convention)
// A, W pre-converted to f16. One wave32 computes a 16x64 tile (1M x 4N WMMA
// tiles, A fragment reused 4x). K templated -> fully unrolled WMMA chain.
// Fragment layouts per CDNA5 ISA 7.12.2 (wave32).
// ---------------------------------------------------------------------------
template <int K_>
__global__ void wmma_gemm_bias(const _Float16* __restrict__ A,
                               const _Float16* __restrict__ W,
                               const float* __restrict__ bias,
                               float* __restrict__ Cout,
                               int M_, int N_) {
    const int lane    = threadIdx.x & 31;
    const int wid     = blockIdx.x * (blockDim.x >> 5) + (threadIdx.x >> 5);
    const int tilesN4 = N_ >> 6;                  // 64-wide column tiles
    const int tilesM  = (M_ + 15) >> 4;
    if (wid >= tilesM * tilesN4) return;          // wave-uniform exit
    const int tm    = wid / tilesN4;
    const int tn4   = wid - tm * tilesN4;
    const int half  = lane >> 4;                  // 0/1 lane group
    const int idx16 = lane & 15;

    const int arow   = tm * 16 + idx16;
    const int arow_c = (arow < M_) ? arow : 0;    // clamp: OOB rows never stored
    const _Float16* Arow = A + (long)arow_c * K_;

    v8f acc[4] = {};
#pragma unroll
    for (int k0 = 0; k0 < K_; k0 += 32) {
        // A 16x32 f16 frag: element i -> K = (i/8)*16 + half*8 + i%8
        const v8h alo = *(const v8h*)(Arow + k0 + half * 8);
        const v8h ahi = *(const v8h*)(Arow + k0 + 16 + half * 8);
        const v16h a = __builtin_shufflevector(alo, ahi,
            0, 1, 2, 3, 4, 5, 6, 7, 8, 9, 10, 11, 12, 13, 14, 15);
#pragma unroll
        for (int j = 0; j < 4; ++j) {
            // B 32x16 f16 frag: element i -> K = (i/8)*16 + half*8 + i%8, N = idx16
            const int bcol = (tn4 * 4 + j) * 16 + idx16;
            const v8h blo = *(const v8h*)(W + (long)bcol * K_ + k0 + half * 8);
            const v8h bhi = *(const v8h*)(W + (long)bcol * K_ + k0 + 16 + half * 8);
            const v16h b = __builtin_shufflevector(blo, bhi,
                0, 1, 2, 3, 4, 5, 6, 7, 8, 9, 10, 11, 12, 13, 14, 15);
            acc[j] = __builtin_amdgcn_wmma_f32_16x16x32_f16(
                false, a, false, b, (short)0, acc[j], false, false);
        }
    }

#pragma unroll
    for (int j = 0; j < 4; ++j) {
#pragma unroll
        for (int v = 0; v < 8; ++v) {
            const int m = tm * 16 + (half << 3) + v;   // C/D: M = v + 8*half
            const int n = (tn4 * 4 + j) * 16 + idx16;  //      N = idx16
            if (m < M_) Cout[(long)m * N_ + n] = acc[j][v] + bias[n];
        }
    }
}

// ---------------------------------------------------------------------------
// Edge kernels (one wave per (edge, head); lane covers 4 channels via float4)
// ---------------------------------------------------------------------------
__global__ void edge_scores(const float* __restrict__ xl, const float* __restrict__ xr,
                            const int* __restrict__ src, const int* __restrict__ dst,
                            const float* __restrict__ att,
                            float* __restrict__ sbuf, float* __restrict__ smax, int E) {
    const int wid  = blockIdx.x * (blockDim.x >> 5) + (threadIdx.x >> 5);
    const int lane = threadIdx.x & 31;
    if (wid >= E * GH) return;
    const int e = wid >> 2, h = wid & 3;
    const int s = src[e], d = dst[e];
    const float4 l4 = *(const float4*)(xl + (long)s * GHC + h * GC + lane * 4);
    const float4 r4 = *(const float4*)(xr + (long)d * GHC + h * GC + lane * 4);
    const float4 a4 = *(const float4*)(att + h * GC + lane * 4);
    float m, acc = 0.f;
    m = l4.x + r4.x; acc += (m > 0.f ? m : m * NEG_SLOPE) * a4.x;
    m = l4.y + r4.y; acc += (m > 0.f ? m : m * NEG_SLOPE) * a4.y;
    m = l4.z + r4.z; acc += (m > 0.f ? m : m * NEG_SLOPE) * a4.z;
    m = l4.w + r4.w; acc += (m > 0.f ? m : m * NEG_SLOPE) * a4.w;
    acc = waveReduceSum(acc);
    if (lane == 0) {
        sbuf[wid] = acc;
        atomicMaxF(&smax[d * GH + h], acc);
    }
}

__global__ void edge_expdenom(const int* __restrict__ dst, float* __restrict__ sbuf,
                              const float* __restrict__ smax, float* __restrict__ denom, int E) {
    const int i = blockIdx.x * blockDim.x + threadIdx.x;
    if (i >= E * GH) return;
    const int e = i >> 2, h = i & 3;
    const int d = dst[e];
    const float v = expf(sbuf[i] - smax[d * GH + h]);
    sbuf[i] = v;
    atomicAdd(&denom[d * GH + h], v);
}

__global__ void edge_aggregate(const float* __restrict__ xl,
                               const int* __restrict__ src, const int* __restrict__ dst,
                               const float* __restrict__ sbuf, const float* __restrict__ denom,
                               float* __restrict__ acc, int E) {
    const int wid  = blockIdx.x * (blockDim.x >> 5) + (threadIdx.x >> 5);
    const int lane = threadIdx.x & 31;
    if (wid >= E * GH) return;
    const int e = wid >> 2, h = wid & 3;
    const int s = src[e], d = dst[e];
    const float alpha = sbuf[wid] / denom[d * GH + h];
    const float4 l4 = *(const float4*)(xl + (long)s * GHC + h * GC + lane * 4);
    float* p = acc + (long)d * GHC + h * GC + lane * 4;
    atomicAdd(p + 0, alpha * l4.x);
    atomicAdd(p + 1, alpha * l4.y);
    atomicAdd(p + 2, alpha * l4.z);
    atomicAdd(p + 3, alpha * l4.w);
}

// mean over heads + bias(es) + ReLU
__global__ void finalize_nodes(const float* __restrict__ acc,
                               const float* __restrict__ b0, const float* __restrict__ b1,
                               float* __restrict__ out, int Nn) {
    const int i = blockIdx.x * blockDim.x + threadIdx.x;
    if (i >= Nn * GC) return;
    const int n = i >> 7, c = i & 127;
    const long base = (long)n * GHC;
    float v = 0.25f * (acc[base + c] + acc[base + GC + c] +
                       acc[base + 2 * GC + c] + acc[base + 3 * GC + c]);
    v += b0[c];
    if (b1) v += b1[c];
    out[i] = v > 0.f ? v : 0.f;
}

__global__ void head_kernel(const float* __restrict__ hs, const float* __restrict__ w,
                            const float* __restrict__ b, float* __restrict__ out, int Nn) {
    const int wid  = blockIdx.x * (blockDim.x >> 5) + (threadIdx.x >> 5);
    const int lane = threadIdx.x & 31;
    if (wid >= Nn) return;
    const float4 x4 = *(const float4*)(hs + (long)wid * GC + lane * 4);
    const float4 w4 = *(const float4*)(w + lane * 4);
    float acc = x4.x * w4.x + x4.y * w4.y + x4.z * w4.z + x4.w * w4.w;
    acc = waveReduceSum(acc);
    if (lane == 0) out[wid] = acc + b[0];
}

__global__ void fill_kernel(float* __restrict__ p, float v, long n) {
    const long i = (long)blockIdx.x * blockDim.x + threadIdx.x;
    if (i < n) p[i] = v;
}

// ---------------------------------------------------------------------------
// Host orchestration
// ---------------------------------------------------------------------------
extern "C" void kernel_launch(void* const* d_in, const int* in_sizes, int n_in,
                              void* d_out, int out_size, void* d_ws, size_t ws_size,
                              hipStream_t stream) {
    (void)n_in; (void)out_size; (void)ws_size;
    const float* x_sample = (const float*)d_in[0];
    const float* x_class  = (const float*)d_in[1];
    const float* pw_s = (const float*)d_in[2];
    const float* pb_s = (const float*)d_in[3];
    const float* pw_c = (const float*)d_in[4];
    const float* pb_c = (const float*)d_in[5];
    const float* wl_all = (const float*)d_in[6];
    const float* bl_all = (const float*)d_in[7];
    const float* wr_all = (const float*)d_in[8];
    const float* br_all = (const float*)d_in[9];
    const float* att_all  = (const float*)d_in[10];
    const float* bias_all = (const float*)d_in[11];
    const float* head_w = (const float*)d_in[12];
    const float* head_b = (const float*)d_in[13];
    const int* src_ss = (const int*)d_in[14];
    const int* dst_ss = (const int*)d_in[15];
    const int* src_cs = (const int*)d_in[16];
    const int* dst_cs = (const int*)d_in[17];
    const int* src_sc = (const int*)d_in[18];
    const int* dst_sc = (const int*)d_in[19];

    const int hid = 128;
    const int Ds = in_sizes[2] / hid;          // proj_sample_w [hid, Ds]
    const int Dc = in_sizes[4] / hid;
    const int N  = in_sizes[0] / Ds;
    const int M  = in_sizes[1] / Dc;
    const int L  = in_sizes[10] / (3 * GH * GC);
    const int E_ss = in_sizes[14], E_cs = in_sizes[16], E_sc = in_sizes[18];
    const int Emax = E_ss > E_cs ? (E_ss > E_sc ? E_ss : E_sc) : (E_cs > E_sc ? E_cs : E_sc);
    const int NMmax = N > M ? N : M;

    // workspace bump allocator (bytes, 256B aligned)
    char* wsb = (char*)d_ws;
    size_t off = 0;
    auto allocB = [&](size_t bytes) {
        char* p = wsb + off;
        off += (bytes + 255) & ~(size_t)255;
        return (void*)p;
    };
    auto allocF = [&](size_t n) { return (float*)allocB(n * sizeof(float)); };
    auto allocH = [&](size_t n) { return (_Float16*)allocB(n * sizeof(_Float16)); };

    float* hsA  = allocF((size_t)N * hid);
    float* hsB  = allocF((size_t)N * hid);
    float* hcA  = allocF((size_t)M * hid);
    float* hcB  = allocF((size_t)M * hid);
    float* xl0  = allocF((size_t)N * GHC);
    float* xr0  = allocF((size_t)N * GHC);
    float* xl1  = allocF((size_t)M * GHC);
    float* xr1  = allocF((size_t)N * GHC);
    float* xl2  = allocF((size_t)N * GHC);
    float* xr2  = allocF((size_t)M * GHC);
    float* accS = allocF((size_t)N * GHC);
    float* accC = allocF((size_t)M * GHC);
    float* sbuf = allocF((size_t)Emax * GH);
    float* smax = allocF((size_t)NMmax * GH);
    float* den  = allocF((size_t)NMmax * GH);
    // f16 operand staging
    _Float16* xs_h  = allocH((size_t)N * Ds);
    _Float16* xc_h  = allocH((size_t)M * Dc);
    _Float16* pws_h = allocH((size_t)hid * Ds);
    _Float16* pwc_h = allocH((size_t)hid * Dc);
    _Float16* wl_h  = allocH((size_t)in_sizes[6]);
    _Float16* wr_h  = allocH((size_t)in_sizes[8]);
    _Float16* hs_h  = allocH((size_t)N * hid);
    _Float16* hc_h  = allocH((size_t)M * hid);

    auto cvt = [&](const float* in, _Float16* outp, long n) {
        const long n4 = n >> 2;  // all our sizes are multiples of 4
        f32_to_f16<<<(int)((n4 + 255) / 256), 256, 0, stream>>>(in, outp, n4);
    };
    auto gemm = [&](const _Float16* A, const _Float16* W, const float* B, float* Co,
                    int M_, int N_, int K_) {
        const int waves = ((M_ + 15) / 16) * (N_ / 64);
        const int blocks = (waves + 7) / 8;
        if (K_ == 256)
            wmma_gemm_bias<256><<<blocks, 256, 0, stream>>>(A, W, B, Co, M_, N_);
        else
            wmma_gemm_bias<128><<<blocks, 256, 0, stream>>>(A, W, B, Co, M_, N_);
    };
    auto fillf = [&](float* p, float v, long n) {
        fill_kernel<<<(int)((n + 255) / 256), 256, 0, stream>>>(p, v, n);
    };
    auto run_edges = [&](const float* xl, const float* xr, const int* src, const int* dst,
                         int E, int n_dst, const float* att, float* acc) {
        const long nh = (long)n_dst * GH;
        fillf(smax, -3.0e38f, nh);
        fillf(den, 0.f, nh);
        const int tasks = E * GH;
        const int wblk = (tasks + 7) / 8;
        edge_scores<<<wblk, 256, 0, stream>>>(xl, xr, src, dst, att, sbuf, smax, E);
        edge_expdenom<<<(tasks + 255) / 256, 256, 0, stream>>>(dst, sbuf, smax, den, E);
        edge_aggregate<<<wblk, 256, 0, stream>>>(xl, src, dst, sbuf, den, acc, E);
    };

    // one-time f16 staging of inputs + all weights
    cvt(x_sample, xs_h, (long)N * Ds);
    cvt(x_class,  xc_h, (long)M * Dc);
    cvt(pw_s, pws_h, (long)hid * Ds);
    cvt(pw_c, pwc_h, (long)hid * Dc);
    cvt(wl_all, wl_h, (long)in_sizes[6]);
    cvt(wr_all, wr_h, (long)in_sizes[8]);

    // input projections
    gemm(xs_h, pws_h, pb_s, hsA, N, hid, Ds);
    gemm(xc_h, pwc_h, pb_c, hcA, M, hid, Dc);

    float* hs = hsA; float* hs_nxt = hsB;
    float* hc = hcA; float* hc_nxt = hcB;

    for (int l = 0; l < L; ++l) {
        const size_t wstep = (size_t)GHC * hid;       // 512*128 per (l,t)
        const size_t bstep = GHC;
        const size_t astep = (size_t)GH * GC;
        const size_t ostep = GC;
        const size_t base = (size_t)l * 3;
        const _Float16* wl[3]; const _Float16* wr[3];
        const float* bl[3]; const float* br[3];
        const float* at[3]; const float* bo[3];
        for (int t = 0; t < 3; ++t) {
            wl[t] = wl_h + (base + t) * wstep;    bl[t] = bl_all + (base + t) * bstep;
            wr[t] = wr_h + (base + t) * wstep;    br[t] = br_all + (base + t) * bstep;
            at[t] = att_all + (base + t) * astep; bo[t] = bias_all + (base + t) * ostep;
        }
        // stage current node features to f16 (hs_h reused by 4 GEMMs, hc_h by 2)
        cvt(hs, hs_h, (long)N * hid);
        cvt(hc, hc_h, (long)M * hid);

        // per-edge-type linear transforms
        gemm(hs_h, wl[0], bl[0], xl0, N, GHC, hid);
        gemm(hs_h, wr[0], br[0], xr0, N, GHC, hid);
        gemm(hc_h, wl[1], bl[1], xl1, M, GHC, hid);
        gemm(hs_h, wr[1], br[1], xr1, N, GHC, hid);
        gemm(hs_h, wl[2], bl[2], xl2, N, GHC, hid);
        gemm(hc_h, wr[2], br[2], xr2, M, GHC, hid);

        fillf(accS, 0.f, (long)N * GHC);
        fillf(accC, 0.f, (long)M * GHC);
        run_edges(xl0, xr0, src_ss, dst_ss, E_ss, N, at[0], accS); // sample->sample
        run_edges(xl1, xr1, src_cs, dst_cs, E_cs, N, at[1], accS); // classifier->sample
        run_edges(xl2, xr2, src_sc, dst_sc, E_sc, M, at[2], accC); // sample->classifier

        finalize_nodes<<<((long)N * GC + 255) / 256, 256, 0, stream>>>(accS, bo[0], bo[1], hs_nxt, N);
        finalize_nodes<<<((long)M * GC + 255) / 256, 256, 0, stream>>>(accC, bo[2], nullptr, hc_nxt, M);

        float* t;
        t = hs; hs = hs_nxt; hs_nxt = t;
        t = hc; hc = hc_nxt; hc_nxt = t;
    }

    head_kernel<<<(N + 7) / 8, 256, 0, stream>>>(hs, head_w, head_b, (float*)d_out, N);
}
